// EncoderLayer_48653389529120
// MI455X (gfx1250) — compile-verified, matrix operand order checked by reading, not run on hardware
//
#include <hip/hip_runtime.h>
#include <hip/hip_bf16.h>

// ---------------------------------------------------------------------------
// MI455X (gfx1250, wave32) transformer encoder layer.
// GEMMs: v_wmma_f32_16x16x32_bf16 fed from LDS tiles staged by the Tensor
// Data Mover (tensor_load_to_lds, TENSORcnt double-buffering).
// Attention: flash-style online softmax, WMMA for QK^T and PV.
// ---------------------------------------------------------------------------

typedef __attribute__((ext_vector_type(16))) __bf16 v16bf;
typedef __attribute__((ext_vector_type(8)))  __bf16 v8bf;
typedef __attribute__((ext_vector_type(8)))  float  v8f;
typedef __attribute__((ext_vector_type(4)))  unsigned int u32x4;
typedef __attribute__((ext_vector_type(8)))  int i32x8;
typedef __attribute__((ext_vector_type(4)))  int i32x4;

#define DM    1024      // d_model
#define DFF   4096      // d_ff
#define SEQ   2048
#define BATCH 4
#define MROWS (BATCH*SEQ)   // 8192
#define NHEAD 16
#define DK    64

static __device__ __forceinline__ v8f wmma_bf16(v16bf a, v16bf b, v8f c) {
    return __builtin_amdgcn_wmma_f32_16x16x32_bf16(
        /*neg_a=*/false, a, /*neg_b=*/false, b,
        /*c_mod=*/(short)0, c, /*reuse_a=*/false, /*reuse_b=*/false);
}

static __device__ __forceinline__ v16bf ldfrag(const __bf16* p0, const __bf16* p1) {
    union { v16bf v; v8bf h[2]; } u;
    u.h[0] = *(const v8bf*)p0;
    u.h[1] = *(const v8bf*)p1;
    return u.v;
}

static __device__ __forceinline__ v8f zero8() {
    v8f z = {0.f,0.f,0.f,0.f,0.f,0.f,0.f,0.f};
    return z;
}

// ---------------------------------------------------------------------------
// TDM: async 2D bf16 tile load  global[rows x tile_k] (row stride = tensor_d0)
// -> LDS, packed rows of tile_k elements. D# layout per CDNA5 ISA ch.8.
// Issued by one wave; completion via TENSORcnt.
// ---------------------------------------------------------------------------
static __device__ __forceinline__ void tdm_load_2d(const __bf16* gptr,
                                                   const __bf16* lptr,
                                                   int tile_k, int rows,
                                                   int tensor_d0, int tensor_d1) {
    const unsigned lds_off = (unsigned)(size_t)lptr;           // LDS generic addr low 32b = LDS byte offset
    const unsigned long long ga = (unsigned long long)(size_t)gptr;
    u32x4 g0;
    g0[0] = 1u;                                                // count=1 (valid), no gather
    g0[1] = lds_off;                                           // lds_addr [63:32]
    g0[2] = (unsigned)(ga & 0xFFFFFFFFu);                      // global_addr [95:64]
    g0[3] = (unsigned)((ga >> 32) & 0x01FFFFFFu) | (2u << 30); // addr[56:32], type=2
    i32x8 g1;
    g1[0] = (int)(1u << 16);                                   // data_size=1 (2 bytes)
    g1[1] = (int)((unsigned)(tensor_d0 & 0xFFFF) << 16);       // tensor_dim0 lo16 @[79:48]
    g1[2] = (int)(((unsigned)tensor_d0 >> 16) |
                  ((unsigned)(tensor_d1 & 0xFFFF) << 16));     // dim0 hi16 | dim1 lo16
    g1[3] = (int)(((unsigned)tensor_d1 >> 16) |
                  ((unsigned)tile_k << 16));                   // dim1 hi16 | tile_dim0
    g1[4] = (int)(unsigned)rows;                               // tile_dim1 (tile_dim2=0)
    g1[5] = (int)(unsigned)tensor_d0;                          // tensor_dim0_stride lo32
    g1[6] = 0;                                                 // stride hi16 | dim1_stride lo16
    g1[7] = 0;
    i32x4 z4 = {0,0,0,0};
#if defined(__clang_major__) && (__clang_major__ >= 23)
    i32x8 z8 = {0,0,0,0,0,0,0,0};
    __builtin_amdgcn_tensor_load_to_lds(g0, g1, z4, z4, z8, 0);
#else
    __builtin_amdgcn_tensor_load_to_lds(g0, g1, z4, z4, 0);
#endif
}

// ---------------------------------------------------------------------------
// fp32 -> bf16 convert
// ---------------------------------------------------------------------------
__global__ __launch_bounds__(256) void k_cvt(const float* __restrict__ src,
                                             __bf16* __restrict__ dst, int n) {
    int i = blockIdx.x * 256 + threadIdx.x;
    if (i < n) dst[i] = (__bf16)src[i];
}

// src is [R,C] fp32 row-major; dst is [C,R] bf16 row-major (transpose).
__global__ __launch_bounds__(256) void k_cvt_t(const float* __restrict__ src,
                                               __bf16* __restrict__ dst,
                                               int R, int C) {
    int i = blockIdx.x * 256 + threadIdx.x;
    if (i < R * C) {
        int r = i / C, c = i % C;
        dst[(size_t)c * R + r] = (__bf16)src[i];
    }
}

// ---------------------------------------------------------------------------
// GEMM: C[M,N] = A[M,K] x B[N,K]^T (bf16 in, f32 accumulate)
// Block 256 thr = 8 waves; block tile 128x128; K stepped by 64.
// A/B K-panels staged in LDS by TDM, double-buffered on TENSORcnt.
// EPI: 0 = store bf16, 1 = store f32, 2 = +bias,ReLU,bf16, 3 = +bias,f32
// ---------------------------------------------------------------------------
template<int EPI>
__global__ __launch_bounds__(256) void k_gemm(const __bf16* __restrict__ A,
                                              const __bf16* __restrict__ B,
                                              const float*  __restrict__ bias,
                                              __bf16* __restrict__ Cb,
                                              float*  __restrict__ Cf,
                                              int M, int N, int K) {
    __shared__ __bf16 lds_a[2][128 * 64];   // 16 KB per buffer
    __shared__ __bf16 lds_b[2][128 * 64];   // 64 KB total of 320 KB WGP LDS

    const int lane = threadIdx.x & 31;
    const int wave = threadIdx.x >> 5;       // 0..7
    const int wm   = wave >> 1;              // 0..3 along M
    const int wn   = wave & 1;               // 0..1 along N
    const int half = lane >> 4;
    const int r    = lane & 15;

    const int mb_blk = blockIdx.x * 128;
    const int nb_blk = blockIdx.y * 128;

    v8f acc[2][4];
#pragma unroll
    for (int i = 0; i < 2; ++i)
#pragma unroll
        for (int j = 0; j < 4; ++j) acc[i][j] = zero8();

    const int nk = K / 64;
    // Prologue: DMA first K-panel (wave 0 only; scalar branch, one TDM pair).
    if (wave == 0) {
        tdm_load_2d(A + (size_t)mb_blk * K, lds_a[0], 64, 128, K, M);
        tdm_load_2d(B + (size_t)nb_blk * K, lds_b[0], 64, 128, K, N);
    }

    for (int it = 0; it < nk; ++it) {
        const int buf = it & 1;
        if (wave == 0) {
            if (it + 1 < nk) {
                const int kb = (it + 1) * 64;
                tdm_load_2d(A + (size_t)mb_blk * K + kb, lds_a[buf ^ 1], 64, 128, K, M);
                tdm_load_2d(B + (size_t)nb_blk * K + kb, lds_b[buf ^ 1], 64, 128, K, N);
                __builtin_amdgcn_s_wait_tensorcnt((short)2);  // pair for 'it' done
            } else {
                __builtin_amdgcn_s_wait_tensorcnt((short)0);
            }
        }
        __syncthreads();   // publish tile 'it' to all waves

#pragma unroll
        for (int ks = 0; ks < 2; ++ks) {     // two 16x16x32 K-substeps
            v16bf a[2], b[4];
#pragma unroll
            for (int i = 0; i < 2; ++i) {
                const __bf16* p = &lds_a[buf][(wm*32 + i*16 + r) * 64 + ks*32 + half*8];
                a[i] = ldfrag(p, p + 16);
            }
#pragma unroll
            for (int j = 0; j < 4; ++j) {
                const __bf16* p = &lds_b[buf][(wn*64 + j*16 + r) * 64 + ks*32 + half*8];
                b[j] = ldfrag(p, p + 16);
            }
#pragma unroll
            for (int i = 0; i < 2; ++i)
#pragma unroll
                for (int j = 0; j < 4; ++j)
                    acc[i][j] = wmma_bf16(a[i], b[j], acc[i][j]);
        }
        __syncthreads();   // all reads done before buffer is re-filled
    }

#pragma unroll
    for (int i = 0; i < 2; ++i) {
#pragma unroll
        for (int j = 0; j < 4; ++j) {
            const int col = nb_blk + wn*64 + j*16 + r;
            float bv = 0.f;
            if (EPI >= 2) bv = bias[col];
#pragma unroll
            for (int e = 0; e < 8; ++e) {
                const int row = mb_blk + wm*32 + i*16 + e + half*8;
                float v = acc[i][j][e];
                if (EPI == 2) { v += bv; v = v > 0.f ? v : 0.f; }
                if (EPI == 3) { v += bv; }
                const size_t idx = (size_t)row * N + col;
                if (EPI == 0 || EPI == 2) Cb[idx] = (__bf16)v;
                else                      Cf[idx] = v;
            }
        }
    }
}

// ---------------------------------------------------------------------------
// Flash attention: grid (S/128, H, B); 8 waves, 16 queries per wave.
// ---------------------------------------------------------------------------
__global__ __launch_bounds__(256) void k_attn(const __bf16* __restrict__ Q,
                                              const __bf16* __restrict__ Kmat,
                                              const __bf16* __restrict__ Vmat,
                                              const int* __restrict__ mask,
                                              __bf16* __restrict__ ctx) {
    __shared__ __bf16 lds_k[32 * DK];        // K tile, row major   (4 KB)
    __shared__ __bf16 lds_vt[DK * 32];       // V tile, transposed  (4 KB)
    __shared__ __bf16 lds_p[8][16 * 32];     // per-wave P tile     (8 KB)

    const int b    = blockIdx.z;
    const int h    = blockIdx.y;
    const int wave = threadIdx.x >> 5;
    const int lane = threadIdx.x & 31;
    const int half = lane >> 4;
    const int r    = lane & 15;
    const int qbase = blockIdx.x * 128 + wave * 16;

    const __bf16* Qh = Q    + (size_t)b * SEQ * DM + (size_t)h * DK;
    const __bf16* Kh = Kmat + (size_t)b * SEQ * DM + (size_t)h * DK;
    const __bf16* Vh = Vmat + (size_t)b * SEQ * DM + (size_t)h * DK;

    v16bf aq[2];
#pragma unroll
    for (int kk = 0; kk < 2; ++kk) {
        const __bf16* p = Qh + (size_t)(qbase + r) * DM + kk*32 + half*8;
        aq[kk] = ldfrag(p, p + 16);
    }

    v8f o[4];
#pragma unroll
    for (int t = 0; t < 4; ++t) o[t] = zero8();
    float mrow[8], lrow[8];
#pragma unroll
    for (int e = 0; e < 8; ++e) { mrow[e] = -3.0e38f; lrow[e] = 0.f; }

    const float scale = 0.125f;  // 1/sqrt(64)

    for (int kb = 0; kb < SEQ; kb += 32) {
        __syncthreads();
        {
            const int t    = threadIdx.x;
            const int krow = t >> 3;
            const int seg  = (t & 7) * 8;
            *(v8bf*)&lds_k[krow * DK + seg] =
                *(const v8bf*)(Kh + (size_t)(kb + krow) * DM + seg);
            v8bf vv = *(const v8bf*)(Vh + (size_t)(kb + krow) * DM + seg);
#pragma unroll
            for (int j = 0; j < 8; ++j)
                lds_vt[(seg + j) * 32 + krow] = vv[j];
        }
        __syncthreads();

        v8f s[2] = { zero8(), zero8() };
#pragma unroll
        for (int kk = 0; kk < 2; ++kk) {
#pragma unroll
            for (int j = 0; j < 2; ++j) {
                const __bf16* p = &lds_k[(j*16 + r) * DK + kk*32 + half*8];
                v16bf bk = ldfrag(p, p + 16);
                s[j] = wmma_bf16(aq[kk], bk, s[j]);
            }
        }

        const int mok0 = mask[(size_t)b * SEQ + kb + r];
        const int mok1 = mask[(size_t)b * SEQ + kb + 16 + r];

        float p0[8], p1[8], alpha[8];
#pragma unroll
        for (int e = 0; e < 8; ++e) {
            float s0 = s[0][e] * scale;
            float s1 = s[1][e] * scale;
            if (mok0 == 0) s0 = -1e9f;
            if (mok1 == 0) s1 = -1e9f;
            float mx = fmaxf(s0, s1);
#pragma unroll
            for (int w = 1; w < 16; w <<= 1)
                mx = fmaxf(mx, __shfl_xor(mx, w, 16));
            const float mnew = fmaxf(mrow[e], mx);
            const float a    = __expf(mrow[e] - mnew);
            const float e0   = __expf(s0 - mnew);
            const float e1   = __expf(s1 - mnew);
            float rs = e0 + e1;
#pragma unroll
            for (int w = 1; w < 16; w <<= 1)
                rs += __shfl_xor(rs, w, 16);
            lrow[e]  = lrow[e] * a + rs;
            mrow[e]  = mnew;
            alpha[e] = a;
            p0[e] = e0;
            p1[e] = e1;
        }
#pragma unroll
        for (int t = 0; t < 4; ++t)
#pragma unroll
            for (int e = 0; e < 8; ++e) o[t][e] *= alpha[e];

        __bf16* lp = lds_p[wave];
#pragma unroll
        for (int e = 0; e < 8; ++e) {
            lp[(e + 8*half) * 32 + r]      = (__bf16)p0[e];
            lp[(e + 8*half) * 32 + 16 + r] = (__bf16)p1[e];
        }
        const __bf16* pp = &lp[r * 32 + half*8];
        v16bf ap = ldfrag(pp, pp + 16);

#pragma unroll
        for (int t = 0; t < 4; ++t) {
            const __bf16* p = &lds_vt[(t*16 + r) * 32 + half*8];
            v16bf bv = ldfrag(p, p + 16);
            o[t] = wmma_bf16(ap, bv, o[t]);
        }
    }

#pragma unroll
    for (int t = 0; t < 4; ++t) {
#pragma unroll
        for (int e = 0; e < 8; ++e) {
            const int row = qbase + e + 8*half;
            const size_t idx = ((size_t)b * SEQ + row) * DM
                             + (size_t)h * DK + t*16 + r;
            ctx[idx] = (__bf16)(o[t][e] / lrow[e]);
        }
    }
}

// ---------------------------------------------------------------------------
// Fused residual + LayerNorm over D=1024. One 256-thread block per row.
// ---------------------------------------------------------------------------
__global__ __launch_bounds__(256) void k_ln(const float* __restrict__ A,
                                            const float* __restrict__ B,
                                            const float* __restrict__ g,
                                            const float* __restrict__ be,
                                            float*  __restrict__ outf,
                                            __bf16* __restrict__ outb) {
    const int row = blockIdx.x;
    const size_t base = (size_t)row * DM;
    float v[4];
    float s = 0.f, s2 = 0.f;
#pragma unroll
    for (int i = 0; i < 4; ++i) {
        const int c = threadIdx.x + i * 256;
        const float t = A[base + c] + B[base + c];
        v[i] = t; s += t; s2 += t * t;
    }
#pragma unroll
    for (int w = 1; w < 32; w <<= 1) {
        s  += __shfl_xor(s,  w, 32);
        s2 += __shfl_xor(s2, w, 32);
    }
    __shared__ float sh[2][8];
    const int wave = threadIdx.x >> 5, lane = threadIdx.x & 31;
    if (lane == 0) { sh[0][wave] = s; sh[1][wave] = s2; }
    __syncthreads();
    float S = 0.f, S2 = 0.f;
#pragma unroll
    for (int i = 0; i < 8; ++i) { S += sh[0][i]; S2 += sh[1][i]; }
    const float mean = S * (1.f / DM);
    const float var  = S2 * (1.f / DM) - mean * mean;
    const float rstd = rsqrtf(var + 1e-5f);
#pragma unroll
    for (int i = 0; i < 4; ++i) {
        const int c = threadIdx.x + i * 256;
        const float ov = (v[i] - mean) * rstd * g[c] + be[c];
        outf[base + c] = ov;
        if (outb) outb[base + c] = (__bf16)ov;
    }
}

// ---------------------------------------------------------------------------
// Host-side orchestration
// ---------------------------------------------------------------------------
extern "C" void kernel_launch(void* const* d_in, const int* in_sizes, int n_in,
                              void* d_out, int out_size, void* d_ws, size_t ws_size,
                              hipStream_t stream) {
    const float* x    = (const float*)d_in[0];
    const int*   mask = (const int*)  d_in[1];
    const float* w_q  = (const float*)d_in[2];
    const float* w_k  = (const float*)d_in[3];
    const float* w_v  = (const float*)d_in[4];
    const float* w_o  = (const float*)d_in[5];
    const float* w1   = (const float*)d_in[6];
    const float* b1   = (const float*)d_in[7];
    const float* w2   = (const float*)d_in[8];
    const float* b2   = (const float*)d_in[9];
    const float* g1   = (const float*)d_in[10];
    const float* be1  = (const float*)d_in[11];
    const float* g2   = (const float*)d_in[12];
    const float* be2  = (const float*)d_in[13];
    float* out = (float*)d_out;

    char* ws = (char*)d_ws;
    size_t off = 0;
    auto carve = [&](size_t bytes) { char* p = ws + off; off += (bytes + 255) & ~size_t(255); return p; };
    __bf16* xb   = (__bf16*)carve((size_t)MROWS * DM * 2);
    __bf16* wqb  = (__bf16*)carve((size_t)DM * DM * 2);
    __bf16* wkb  = (__bf16*)carve((size_t)DM * DM * 2);
    __bf16* wvb  = (__bf16*)carve((size_t)DM * DM * 2);
    __bf16* wob  = (__bf16*)carve((size_t)DM * DM * 2);
    __bf16* w1tb = (__bf16*)carve((size_t)DFF * DM * 2);   // [DFF, DM]
    __bf16* w2tb = (__bf16*)carve((size_t)DM * DFF * 2);   // [DM, DFF]
    __bf16* Qb   = (__bf16*)carve((size_t)MROWS * DM * 2);
    __bf16* Kb   = (__bf16*)carve((size_t)MROWS * DM * 2);
    __bf16* Vb   = (__bf16*)carve((size_t)MROWS * DM * 2);
    __bf16* ctxb = (__bf16*)carve((size_t)MROWS * DM * 2);
    float*  attnf= (float*) carve((size_t)MROWS * DM * 4);
    float*  hf   = (float*) carve((size_t)MROWS * DM * 4);
    __bf16* hb   = (__bf16*)carve((size_t)MROWS * DM * 2);
    __bf16* ff1b = (__bf16*)carve((size_t)MROWS * DFF * 2);
    float*  ff2f = (float*) carve((size_t)MROWS * DM * 4);
    if (off > ws_size) return;

    {
        const int nx = MROWS * DM;
        k_cvt<<<(nx + 255) / 256, 256, 0, stream>>>(x, xb, nx);
        const int nw = DM * DM;
        k_cvt<<<(nw + 255) / 256, 256, 0, stream>>>(w_q, wqb, nw);
        k_cvt<<<(nw + 255) / 256, 256, 0, stream>>>(w_k, wkb, nw);
        k_cvt<<<(nw + 255) / 256, 256, 0, stream>>>(w_v, wvb, nw);
        k_cvt<<<(nw + 255) / 256, 256, 0, stream>>>(w_o, wob, nw);
        const int nf = DM * DFF;
        k_cvt_t<<<(nf + 255) / 256, 256, 0, stream>>>(w1, w1tb, DM, DFF);  // -> [DFF,DM]
        k_cvt_t<<<(nf + 255) / 256, 256, 0, stream>>>(w2, w2tb, DFF, DM);  // -> [DM,DFF]
    }

    {
        dim3 grid(MROWS / 128, DM / 128);
        k_gemm<0><<<grid, 256, 0, stream>>>(xb, wqb, nullptr, Qb, nullptr, MROWS, DM, DM);
        k_gemm<0><<<grid, 256, 0, stream>>>(xb, wkb, nullptr, Kb, nullptr, MROWS, DM, DM);
        k_gemm<0><<<grid, 256, 0, stream>>>(xb, wvb, nullptr, Vb, nullptr, MROWS, DM, DM);
    }

    {
        dim3 grid(SEQ / 128, NHEAD, BATCH);
        k_attn<<<grid, 256, 0, stream>>>(Qb, Kb, Vb, mask, ctxb);
    }

    {
        dim3 grid(MROWS / 128, DM / 128);
        k_gemm<1><<<grid, 256, 0, stream>>>(ctxb, wob, nullptr, nullptr, attnf, MROWS, DM, DM);
    }

    k_ln<<<MROWS, 256, 0, stream>>>(x, attnf, g1, be1, hf, hb);

    {
        dim3 grid(MROWS / 128, DFF / 128);
        k_gemm<2><<<grid, 256, 0, stream>>>(hb, w1tb, b1, ff1b, nullptr, MROWS, DFF, DM);
    }

    {
        dim3 grid(MROWS / 128, DM / 128);
        k_gemm<3><<<grid, 256, 0, stream>>>(ff1b, w2tb, b2, nullptr, ff2f, MROWS, DM, DFF);
    }

    k_ln<<<MROWS, 256, 0, stream>>>(hf, ff2f, g2, be2, out, nullptr);
}
